// SelfAttention_18133351924118
// MI455X (gfx1250) — matrix operand order repaired and checked by reading, not compiled, not run on hardware
//
#include <hip/hip_runtime.h>

// ---------------------------------------------------------------------------
// Self-attention block for MI455X (gfx1250): bf16 WMMA everywhere.
// Copy-free ping-pong software pipelining (no rotating-buffer register moves).
// ---------------------------------------------------------------------------

typedef __attribute__((ext_vector_type(16))) __bf16 v16bf;
typedef __attribute__((ext_vector_type(8)))  __bf16 v8bf;
typedef __attribute__((ext_vector_type(8)))  float  v8f;

#define B_    2
#define S_    2048
#define D_    1024
#define H_    16
#define HD_   64
#define M_TOT (B_ * S_)   // 4096
#define N_QKV (3 * D_)    // 3072

// ---- WMMA fragment helpers (contiguous k = 16*half + e in A and B, wave32) ----------

// A (16x32 bf16): lane l -> row l%16; elems e -> K = e + 16*(l>=16)
static __device__ __forceinline__ v16bf load_a_frag(const __bf16* base, int ldk) {
  const int lane = threadIdx.x & 31;
  const int half = lane >> 4;
  const int r    = lane & 15;
  const __bf16* p = base + r * ldk + 16 * half;
  return *(const v16bf*)(p);
}

// B (32x16 bf16), source stored transposed [n][k]: lane l -> col l%16,
// elems e -> K = e + 16*(l>=16)  => one contiguous 32B load per lane.
static __device__ __forceinline__ v16bf load_b_frag(const __bf16* base, int ldk) {
  const int lane = threadIdx.x & 31;
  const int half = lane >> 4;
  const int n    = lane & 15;
  return *(const v16bf*)(base + n * ldk + 16 * half);
}

static __device__ __forceinline__ v8f wmma_bf16(v16bf a, v16bf b, v8f c) {
  return __builtin_amdgcn_wmma_f32_16x16x32_bf16(
      /*neg_a=*/false, a, /*neg_b=*/false, b,
      /*c_mod=*/(short)0, c, /*reuse_a=*/false, /*reuse_b=*/false);
}

// Fragment set for one 32x64 x K-step: 2 A-frags + 4 B-frags.
struct Frags {
  v16bf a0, a1;
  v16bf b[4];
};

static __device__ __forceinline__ void load_frags(Frags& f,
                                                  const __bf16* arow0,
                                                  const __bf16* arow1,
                                                  const __bf16* wTn0, int k) {
  f.a0 = load_a_frag(arow0 + k, D_);
  f.a1 = load_a_frag(arow1 + k, D_);
#pragma unroll
  for (int t = 0; t < 4; ++t)
    f.b[t] = load_b_frag(wTn0 + (size_t)(16 * t) * D_ + k, D_);
}

static __device__ __forceinline__ void mma_frags(v8f acc[2][4], const Frags& f) {
#pragma unroll
  for (int t = 0; t < 4; ++t) acc[0][t] = wmma_bf16(f.a0, f.b[t], acc[0][t]);
#pragma unroll
  for (int t = 0; t < 4; ++t) acc[1][t] = wmma_bf16(f.a1, f.b[t], acc[1][t]);
}

// ---- precision conversion / weight transpose -------------------------------

__global__ void k_cvt_bf16(const float* __restrict__ src, __bf16* __restrict__ dst, int n) {
  int i = blockIdx.x * blockDim.x + threadIdx.x;
  if (i < n) dst[i] = (__bf16)src[i];
}

// dst[n*K + k] = bf16(src[k*N + n])  (weights are small; L2-resident)
__global__ void k_cvt_transpose(const float* __restrict__ src, __bf16* __restrict__ dst,
                                int K, int N) {
  int i = blockIdx.x * blockDim.x + threadIdx.x;
  if (i < K * N) {
    int n = i / K, k = i % K;
    dst[i] = (__bf16)src[k * N + n];
  }
}

// ---- GEMM 1: qkv = x @ Wqkv + b; scatter into Q,K (row-major) and V^T ------
// Wave tile: 32 (M) x 64 (N); ping-pong double buffering, K unrolled by 2.

__global__ void k_gemm_qkv(const __bf16* __restrict__ xb,
                           const __bf16* __restrict__ wT,
                           const float*  __restrict__ bias,
                           __bf16* __restrict__ qbuf,
                           __bf16* __restrict__ kbuf,
                           __bf16* __restrict__ vtbuf) {
  const int wave = threadIdx.x >> 5;
  const int lane = threadIdx.x & 31;
  const int half = lane >> 4;
  const int n_l  = lane & 15;
  const int m0   = blockIdx.y * 32;
  const int n0   = blockIdx.x * 256 + wave * 64;

  const __bf16* arow0 = xb + (size_t)m0 * D_;
  const __bf16* arow1 = xb + (size_t)(m0 + 16) * D_;
  const __bf16* wTn0  = wT + (size_t)n0 * D_;

  v8f acc[2][4] = {};
  Frags fA, fB;
  load_frags(fA, arow0, arow1, wTn0, 0);

  for (int k0 = 0; k0 < D_; k0 += 64) {
    const int kpf = (k0 + 256 < D_) ? (k0 + 256) : (D_ - 32);
    __builtin_prefetch(arow0 + kpf, 0, 3);
    __builtin_prefetch(wTn0 + kpf, 0, 3);
    load_frags(fB, arow0, arow1, wTn0, k0 + 32);    // k0+32 < D_ always
    mma_frags(acc, fA);
    const int k2 = (k0 + 64 < D_) ? (k0 + 64) : 0;  // last iter: dummy reload
    load_frags(fA, arow0, arow1, wTn0, k2);
    mma_frags(acc, fB);
  }

#pragma unroll
  for (int u = 0; u < 2; ++u) {
#pragma unroll
    for (int t = 0; t < 4; ++t) {
#pragma unroll
      for (int i = 0; i < 8; ++i) {
        int m = i + 8 * half;
        int r = m0 + 16 * u + m;        // [0, 4096)
        int c = n0 + 16 * t + n_l;      // [0, 3072)
        float v = acc[u][t][i] + bias[c];
        int part = c >> 10;             // 0:q 1:k 2:v
        int cc = c & 1023;
        int h = cc >> 6, d = cc & 63;
        int bi = r >> 11, s = r & 2047;
        int bh = bi * H_ + h;
        __bf16 bv = (__bf16)v;
        if (part == 0)      qbuf[(size_t)(bh * S_ + s) * HD_ + d] = bv;
        else if (part == 1) kbuf[(size_t)(bh * S_ + s) * HD_ + d] = bv;
        else                vtbuf[(size_t)(bh * HD_ + d) * S_ + s] = bv;
      }
    }
  }
}

// ---- Attention helpers -----------------------------------------------------

static __device__ __forceinline__ void load_kfrags(v16bf kb[4], const __bf16* kbase, int j) {
#pragma unroll
  for (int t = 0; t < 2; ++t) {
    const __bf16* p = kbase + (size_t)(j + 16 * t) * HD_;
    kb[2 * t]     = load_b_frag(p, HD_);
    kb[2 * t + 1] = load_b_frag(p + 32, HD_);
  }
}

// One 32-key score tile -> update running row max / row sum.
static __device__ __forceinline__ void score_stats(const v16bf& qa0, const v16bf& qa1,
                                                   const v16bf kb[4],
                                                   float mrow[8], float lrow[8]) {
  const float scale = 0.125f;
  v8f sc[2] = {};
  sc[0] = wmma_bf16(qa0, kb[0], sc[0]);
  sc[0] = wmma_bf16(qa1, kb[1], sc[0]);
  sc[1] = wmma_bf16(qa0, kb[2], sc[1]);
  sc[1] = wmma_bf16(qa1, kb[3], sc[1]);
#pragma unroll
  for (int i = 0; i < 8; ++i) {
    float s0 = sc[0][i] * scale, s1 = sc[1][i] * scale;
    float tm = fmaxf(s0, s1);
#pragma unroll
    for (int msk = 1; msk < 16; msk <<= 1) tm = fmaxf(tm, __shfl_xor(tm, msk, 32));
    float mnew = fmaxf(mrow[i], tm);
    float ts = __expf(s0 - mnew) + __expf(s1 - mnew);
#pragma unroll
    for (int msk = 1; msk < 16; msk <<= 1) ts += __shfl_xor(ts, msk, 32);
    lrow[i] = lrow[i] * __expf(mrow[i] - mnew) + ts;
    mrow[i] = mnew;
  }
}

// ---- Attention: one wave per (b,h,16-query tile); two-pass exact softmax ---

__global__ void k_attention(const __bf16* __restrict__ qbuf,
                            const __bf16* __restrict__ kbuf,
                            const __bf16* __restrict__ vtbuf,
                            float* __restrict__ attn_out,   // [B,H,S,S]
                            __bf16* __restrict__ obuf) {    // [B*S, D]
  __shared__ __align__(64) __bf16 ldsP[4][16 * 32];
  const int wave = threadIdx.x >> 5;
  const int lane = threadIdx.x & 31;
  const int half = lane >> 4;
  const int n_l  = lane & 15;
  const int wid  = blockIdx.x * 4 + wave;
  const int qtile = wid & 127;          // S/16 tiles
  const int bh    = wid >> 7;           // [0, 32)
  const int q0    = qtile * 16;
  const float scale = 0.125f;           // 1/sqrt(HD)

  const __bf16* qp    = qbuf  + (size_t)(bh * S_ + q0) * HD_;
  const __bf16* kbase = kbuf  + (size_t)bh * S_ * HD_;
  const __bf16* vbase = vtbuf + (size_t)bh * HD_ * S_;
  v16bf qa0 = load_a_frag(qp, HD_);       // K 0..31  of HD
  v16bf qa1 = load_a_frag(qp + 32, HD_);  // K 32..63

  float mrow[8], lrow[8];
#pragma unroll
  for (int i = 0; i < 8; ++i) { mrow[i] = -3.0e38f; lrow[i] = 0.f; }

  // ---- pass 1: streaming row max / row sum; ping-pong K fragments ----
  {
    v16bf kbA[4], kbB[4];
    load_kfrags(kbA, kbase, 0);
    for (int j = 0; j < S_; j += 64) {
      load_kfrags(kbB, kbase, j + 32);              // j+32 < S_ always
      score_stats(qa0, qa1, kbA, mrow, lrow);
      const int j2 = (j + 64 < S_) ? (j + 64) : 0;  // last iter: dummy reload
      load_kfrags(kbA, kbase, j2);
      score_stats(qa0, qa1, kbB, mrow, lrow);
    }
  }
  float linv[8];
#pragma unroll
  for (int i = 0; i < 8; ++i) linv[i] = 1.f / lrow[i];

  // ---- pass 2: write probs, accumulate O = P @ V via WMMA ----
  v8f oacc[4] = {};
  __bf16* pbuf = &ldsP[wave][0];
  for (int j = 0; j < S_; j += 32) {
    v8f sc[2] = {};
#pragma unroll
    for (int t = 0; t < 2; ++t) {
      const __bf16* kp2 = kbase + (size_t)(j + 16 * t) * HD_;
      sc[t] = wmma_bf16(qa0, load_b_frag(kp2, HD_), sc[t]);
      sc[t] = wmma_bf16(qa1, load_b_frag(kp2 + 32, HD_), sc[t]);
    }
#pragma unroll
    for (int t = 0; t < 2; ++t) {
#pragma unroll
      for (int i = 0; i < 8; ++i) {
        int m = i + 8 * half;
        float p = __expf(sc[t][i] * scale - mrow[i]) * linv[i];
        attn_out[(size_t)(bh * S_ + q0 + m) * S_ + j + 16 * t + n_l] = p;
        pbuf[m * 32 + 16 * t + n_l] = (__bf16)p;   // C-layout -> LDS row-major
      }
    }
    __builtin_amdgcn_wave_barrier();               // DS ops are wave-ordered
    v16bf pa = load_a_frag(pbuf, 32);              // re-read as A fragment
    __builtin_amdgcn_wave_barrier();
#pragma unroll
    for (int t = 0; t < 4; ++t) {
      v16bf vb = load_b_frag(vbase + (size_t)(16 * t) * S_ + j, S_);
      oacc[t] = wmma_bf16(pa, vb, oacc[t]);
    }
  }

  const int bi = bh >> 4, h = bh & 15;
#pragma unroll
  for (int t = 0; t < 4; ++t) {
#pragma unroll
    for (int i = 0; i < 8; ++i) {
      int m = i + 8 * half;
      obuf[(size_t)(bi * S_ + q0 + m) * D_ + h * HD_ + 16 * t + n_l] = (__bf16)oacc[t][i];
    }
  }
}

// ---- GEMM 2: y = o @ Wout + b_out + x (residual), fp32 out -----------------

__global__ void k_gemm_out(const __bf16* __restrict__ ob,
                           const __bf16* __restrict__ wT,
                           const float*  __restrict__ bias,
                           const float*  __restrict__ xres,
                           float* __restrict__ y) {
  const int wave = threadIdx.x >> 5;
  const int lane = threadIdx.x & 31;
  const int half = lane >> 4;
  const int n_l  = lane & 15;
  const int m0   = blockIdx.y * 32;
  const int n0   = blockIdx.x * 256 + wave * 64;

  const __bf16* arow0 = ob + (size_t)m0 * D_;
  const __bf16* arow1 = ob + (size_t)(m0 + 16) * D_;
  const __bf16* wTn0  = wT + (size_t)n0 * D_;

  v8f acc[2][4] = {};
  Frags fA, fB;
  load_frags(fA, arow0, arow1, wTn0, 0);

  for (int k0 = 0; k0 < D_; k0 += 64) {
    const int kpf = (k0 + 256 < D_) ? (k0 + 256) : (D_ - 32);
    __builtin_prefetch(arow0 + kpf, 0, 3);
    __builtin_prefetch(wTn0 + kpf, 0, 3);
    load_frags(fB, arow0, arow1, wTn0, k0 + 32);
    mma_frags(acc, fA);
    const int k2 = (k0 + 64 < D_) ? (k0 + 64) : 0;
    load_frags(fA, arow0, arow1, wTn0, k2);
    mma_frags(acc, fB);
  }

#pragma unroll
  for (int u = 0; u < 2; ++u) {
#pragma unroll
    for (int t = 0; t < 4; ++t) {
#pragma unroll
      for (int i = 0; i < 8; ++i) {
        int m = i + 8 * half;
        int r = m0 + 16 * u + m;
        int c = n0 + 16 * t + n_l;
        y[(size_t)r * D_ + c] = acc[u][t][i] + bias[c] + xres[(size_t)r * D_ + c];
      }
    }
  }
}

// ---- LayerNorm over D=1024, one block (256 thr) per row --------------------

__global__ void k_layernorm(const float* __restrict__ y,
                            const float* __restrict__ g,
                            const float* __restrict__ bta,
                            float* __restrict__ out) {
  __shared__ float red[256];
  const int r = blockIdx.x;
  const int tid = threadIdx.x;
  float vals[4];
  float s = 0.f;
#pragma unroll
  for (int j = 0; j < 4; ++j) { vals[j] = y[(size_t)r * D_ + tid + 256 * j]; s += vals[j]; }
  red[tid] = s; __syncthreads();
  for (int o = 128; o > 0; o >>= 1) { if (tid < o) red[tid] += red[tid + o]; __syncthreads(); }
  float mu = red[0] / (float)D_;
  __syncthreads();
  float s2 = 0.f;
#pragma unroll
  for (int j = 0; j < 4; ++j) { float d = vals[j] - mu; s2 += d * d; }
  red[tid] = s2; __syncthreads();
  for (int o = 128; o > 0; o >>= 1) { if (tid < o) red[tid] += red[tid + o]; __syncthreads(); }
  float var = red[0] / (float)D_;
  float rstd = rsqrtf(var + 1e-5f);
#pragma unroll
  for (int j = 0; j < 4; ++j) {
    int c = tid + 256 * j;
    out[(size_t)r * D_ + c] = (vals[j] - mu) * rstd * g[c] + bta[c];
  }
}

// ---------------------------------------------------------------------------

extern "C" void kernel_launch(void* const* d_in, const int* in_sizes, int n_in,
                              void* d_out, int out_size, void* d_ws, size_t ws_size,
                              hipStream_t stream) {
  (void)in_sizes; (void)n_in; (void)out_size; (void)ws_size;
  const float* x     = (const float*)d_in[0];
  const float* w_qkv = (const float*)d_in[1];
  const float* b_qkv = (const float*)d_in[2];
  const float* w_out = (const float*)d_in[3];
  const float* b_out = (const float*)d_in[4];
  const float* ln_g  = (const float*)d_in[5];
  const float* ln_b  = (const float*)d_in[6];

  float* out  = (float*)d_out;                     // [B,S,D]
  float* attn = out + (size_t)M_TOT * D_;          // [B,H,S,S]

  char* ws = (char*)d_ws;
  size_t off = 0;
  auto salloc = [&](size_t bytes) -> char* {
    char* p = ws + off;
    off += (bytes + 255) & ~(size_t)255;
    return p;
  };
  __bf16* xb    = (__bf16*)salloc((size_t)M_TOT * D_ * 2);          // 8 MB
  __bf16* wqkvT = (__bf16*)salloc((size_t)N_QKV * D_ * 2);          // 6 MB
  __bf16* woutT = (__bf16*)salloc((size_t)D_ * D_ * 2);             // 2 MB
  __bf16* qbuf  = (__bf16*)salloc((size_t)B_ * H_ * S_ * HD_ * 2);  // 8 MB
  __bf16* kbuf  = (__bf16*)salloc((size_t)B_ * H_ * S_ * HD_ * 2);  // 8 MB
  __bf16* vtbuf = (__bf16*)salloc((size_t)B_ * H_ * S_ * HD_ * 2);  // 8 MB
  __bf16* obuf  = (__bf16*)salloc((size_t)M_TOT * D_ * 2);          // 8 MB
  float*  ybuf  = (float*) salloc((size_t)M_TOT * D_ * 4);          // 16 MB

  const int n1 = M_TOT * D_;
  k_cvt_bf16<<<(n1 + 255) / 256, 256, 0, stream>>>(x, xb, n1);
  k_cvt_transpose<<<(N_QKV * D_ + 255) / 256, 256, 0, stream>>>(w_qkv, wqkvT, D_, N_QKV);
  k_cvt_transpose<<<(D_ * D_ + 255) / 256, 256, 0, stream>>>(w_out, woutT, D_, D_);

  dim3 g1(N_QKV / 256, M_TOT / 32);
  k_gemm_qkv<<<g1, 128, 0, stream>>>(xb, wqkvT, b_qkv, qbuf, kbuf, vtbuf);

  k_attention<<<(B_ * H_ * (S_ / 16)) / 4, 128, 0, stream>>>(qbuf, kbuf, vtbuf, attn, obuf);

  dim3 g2(D_ / 256, M_TOT / 32);
  k_gemm_out<<<g2, 128, 0, stream>>>(obuf, woutT, b_out, x, ybuf);

  k_layernorm<<<M_TOT, 256, 0, stream>>>(ybuf, ln_g, ln_b, out);
}
